// CustomLSTMCell_28664611733558
// MI455X (gfx1250) — compile-verified, matrix operand order checked by reading, not run on hardware
//
#include <hip/hip_runtime.h>
#include <hip/hip_bf16.h>

typedef __attribute__((ext_vector_type(16))) __bf16 v16bf;
typedef __attribute__((ext_vector_type(8)))  float  v8f;

#define Bsz  8192
#define INsz 512
#define Hsz  1024
#define BM   128
#define BN   64
#define BK   32
#define THREADS 256

// LDS layout (dynamic, base offset 0), per buffer:
//   A tile [128][32] bf16            = 8192 B   at +0
//   B tiles [4][64][32] bf16         = 16384 B  at +8192
#define LDS_B_OFF      8192u
#define LDS_BUF_STRIDE 24576u
#define LDS_TOTAL      (2 * LDS_BUF_STRIDE)

__device__ __forceinline__ float sigmoidf_(float v) {
    return 1.0f / (1.0f + __expf(-v));
}

// CDNA5 async copy: per-lane 16B global -> LDS, tracked by ASYNCcnt
__device__ __forceinline__ void async_ld_b128(unsigned lds_off, const void* gaddr) {
    asm volatile("global_load_async_to_lds_b128 %0, %1, off"
                 :: "v"(lds_off), "v"(gaddr)
                 : "memory");
}
__device__ __forceinline__ void wait_asynccnt0() {
    asm volatile("s_wait_asynccnt 0x0" ::: "memory");
}

// ---------------- conversion pre-pass: f32 -> bf16 ----------------------------
__global__ __launch_bounds__(256)
void cvt_f32_to_bf16(const float* __restrict__ src, __bf16* __restrict__ dst, int n4) {
    int i = blockIdx.x * blockDim.x + threadIdx.x;
    if (i < n4) {
        const float4 v = ((const float4*)src)[i];
        __bf16* d = dst + (size_t)i * 4;
        d[0] = (__bf16)v.x; d[1] = (__bf16)v.y; d[2] = (__bf16)v.z; d[3] = (__bf16)v.w;
    }
}

// ---------------- stage one K-step's tiles via async DMA ----------------------
// A tile: 128 rows x 64 B  = 512  x 16B chunks (2 per thread)
// B tile: 4 x 64 rows x 64 B = 1024 x 16B chunks (4 per thread)
__device__ __forceinline__ void stage_tiles(
    const __bf16* __restrict__ Abase, const __bf16* __restrict__ Wbase,
    size_t gstride, int ldA, int m0, int n0, int k0,
    unsigned ldsbase, int tid)
{
    #pragma unroll
    for (int i = 0; i < 2; ++i) {
        const int idx = tid + i * THREADS;        // 0..511
        const int row = idx >> 2;
        const int kc  = idx & 3;
        const void* g = Abase + (size_t)(m0 + row) * ldA + k0 + kc * 8;
        async_ld_b128(ldsbase + (unsigned)(row * 64 + kc * 16), g);
    }
    #pragma unroll
    for (int i = 0; i < 4; ++i) {
        const int idx = tid + i * THREADS;        // 0..1023
        const int gg  = idx >> 8;                  // gate
        const int rem = idx & 255;
        const int n   = rem >> 2;
        const int kc  = rem & 3;
        const void* g = Wbase + (size_t)gg * gstride
                              + (size_t)(n0 + n) * ldA + k0 + kc * 8;
        async_ld_b128(ldsbase + LDS_B_OFF + (unsigned)(gg * 4096 + n * 64 + kc * 16), g);
    }
}

// ---------------- fused GEMM (4 gates) + LSTM epilogue ------------------------
__global__ __launch_bounds__(THREADS)
void lstm_wmma_kernel(
    const __bf16* __restrict__ xb, const __bf16* __restrict__ hb,
    const __bf16* __restrict__ wb, const __bf16* __restrict__ ub,   // [4][H][K]
    const float* __restrict__ c_prev, const float* __restrict__ boundary,
    const float* __restrict__ b_Wi, const float* __restrict__ b_Ui,
    const float* __restrict__ b_Wf, const float* __restrict__ b_Uf,
    const float* __restrict__ b_Wo, const float* __restrict__ b_Uo,
    const float* __restrict__ b_Wg, const float* __restrict__ b_Ug,
    const float* __restrict__ W_b,  const float* __restrict__ b_Wb,
    float* __restrict__ out_h, float* __restrict__ out_c)
{
    extern __shared__ char smem[];

    const int tid  = threadIdx.x;
    const int lane = tid & 31;
    const int wave = tid >> 5;
    const int wm   = wave >> 1;   // 0..3  : 32-row slab
    const int wn   = wave & 1;    // 0..1  : 32-col slab

    const int m0 = blockIdx.x * BM;
    const int n0 = blockIdx.y * BN;

    // per-lane output columns for the two N-subtiles
    int   ncol[2];
    float bias_i[2], bias_f[2], bias_o[2], bias_g[2], wb0[2], wb1[2];
    #pragma unroll
    for (int t = 0; t < 2; ++t) {
        const int nc = n0 + wn * 32 + t * 16 + (lane & 15);
        ncol[t]   = nc;
        bias_i[t] = b_Wi[nc] + b_Ui[nc];
        bias_f[t] = b_Wf[nc] + b_Uf[nc] + b_Wb[nc];
        bias_o[t] = b_Wo[nc] + b_Uo[nc];
        bias_g[t] = b_Wg[nc] + b_Ug[nc];
        wb0[t]    = W_b[nc * 2 + 0];
        wb1[t]    = W_b[nc * 2 + 1];
    }

    v8f acc[4][2][2] = {};   // [gate][m-subtile][n-subtile]

    const int    KT1 = INsz / BK;                 // 16
    const int    KT2 = Hsz  / BK;                 // 32
    const int    TOT = KT1 + KT2;                 // 48
    const size_t gsW = (size_t)Hsz * INsz;
    const size_t gsU = (size_t)Hsz * Hsz;

    // prologue: stage K-step 0 into buffer 0
    stage_tiles(xb, wb, gsW, INsz, m0, n0, 0, 0u, tid);

    for (int kt = 0; kt < TOT; ++kt) {
        const unsigned cur = (unsigned)(kt & 1) * LDS_BUF_STRIDE;

        wait_asynccnt0();      // own async writes for `cur` complete
        __syncthreads();       // all waves' writes landed; other buffer free

        if (kt + 1 < TOT) {
            const bool p1  = (kt + 1 < KT1);
            const int  k0n = (p1 ? (kt + 1) : (kt + 1 - KT1)) * BK;
            stage_tiles(p1 ? xb : hb, p1 ? wb : ub, p1 ? gsW : gsU,
                        p1 ? INsz : Hsz, m0, n0, k0n,
                        LDS_BUF_STRIDE - cur, tid);
        }

        // fragments: lane halves split K; low nibble = M (A) / N (B)
        const unsigned khB = (unsigned)(lane >> 4) * 32u;
        const int      lr  = lane & 15;
        const char* a_ptr = smem + cur + (unsigned)(wm * 32 + lr) * 64u + khB;
        const char* b_ptr = smem + cur + LDS_B_OFF + (unsigned)(wn * 32 + lr) * 64u + khB;

        v16bf afrag[2];
        afrag[0] = *(const v16bf*)(a_ptr);
        afrag[1] = *(const v16bf*)(a_ptr + 16u * 64u);

        v16bf bfrag[4][2];
        #pragma unroll
        for (int g = 0; g < 4; ++g)
            #pragma unroll
            for (int t = 0; t < 2; ++t)
                bfrag[g][t] = *(const v16bf*)(b_ptr + (unsigned)g * 4096u + (unsigned)t * 1024u);

        #pragma unroll
        for (int g = 0; g < 4; ++g)
            #pragma unroll
            for (int s = 0; s < 2; ++s)
                #pragma unroll
                for (int t = 0; t < 2; ++t)
                    acc[g][s][t] = __builtin_amdgcn_wmma_f32_16x16x32_bf16(
                        false, afrag[s], false, bfrag[g][t], (short)0, acc[g][s][t], false, false);
    }

    // ---- fused epilogue ----
    const int mbase = m0 + wm * 32;
    #pragma unroll
    for (int s = 0; s < 2; ++s) {
        #pragma unroll
        for (int r = 0; r < 8; ++r) {
            const int   m  = mbase + s * 16 + ((lane < 16) ? r : r + 8);
            const float bd0 = boundary[2 * m];
            const float bd1 = boundary[2 * m + 1];
            #pragma unroll
            for (int t = 0; t < 2; ++t) {
                const float ai = acc[0][s][t][r];
                const float af = acc[1][s][t][r];
                const float ao = acc[2][s][t][r];
                const float ag = acc[3][s][t][r];

                const float binf = bd0 * wb0[t] + bd1 * wb1[t];

                const float iv = sigmoidf_(ai + bias_i[t]);
                const float fv = sigmoidf_(af + bias_f[t] + binf);
                const float ov = sigmoidf_(ao + bias_o[t]);
                const float gv = tanhf(ag + bias_g[t]);

                const size_t off = (size_t)m * Hsz + ncol[t];
                const float cv = fv * c_prev[off] + iv * gv;
                const float hv = ov * tanhf(cv);
                out_h[off] = hv;
                out_c[off] = cv;
            }
        }
    }
}

extern "C" void kernel_launch(void* const* d_in, const int* in_sizes, int n_in,
                              void* d_out, int out_size, void* d_ws, size_t ws_size,
                              hipStream_t stream) {
    (void)in_sizes; (void)n_in; (void)out_size; (void)ws_size;

    const float* x        = (const float*)d_in[0];
    const float* h_prev   = (const float*)d_in[1];
    const float* c_prev   = (const float*)d_in[2];
    const float* boundary = (const float*)d_in[3];
    const float* W_i  = (const float*)d_in[4];
    const float* b_Wi = (const float*)d_in[5];
    const float* U_i  = (const float*)d_in[6];
    const float* b_Ui = (const float*)d_in[7];
    const float* W_f  = (const float*)d_in[8];
    const float* b_Wf = (const float*)d_in[9];
    const float* U_f  = (const float*)d_in[10];
    const float* b_Uf = (const float*)d_in[11];
    const float* W_o  = (const float*)d_in[12];
    const float* b_Wo = (const float*)d_in[13];
    const float* U_o  = (const float*)d_in[14];
    const float* b_Uo = (const float*)d_in[15];
    const float* W_g  = (const float*)d_in[16];
    const float* b_Wg = (const float*)d_in[17];
    const float* U_g  = (const float*)d_in[18];
    const float* b_Ug = (const float*)d_in[19];
    const float* W_b  = (const float*)d_in[20];
    const float* b_Wb = (const float*)d_in[21];

    float* out_h = (float*)d_out;
    float* out_c = out_h + (size_t)Bsz * Hsz;

    // ---- bf16 workspace layout ----
    __bf16* ws   = (__bf16*)d_ws;
    const size_t nx = (size_t)Bsz * INsz;
    const size_t nh = (size_t)Bsz * Hsz;
    const size_t nW = (size_t)Hsz * INsz;
    const size_t nU = (size_t)Hsz * Hsz;
    __bf16* xb  = ws;
    __bf16* hb  = xb + nx;
    __bf16* wbp = hb + nh;                          // [4][H][IN]
    __bf16* ubp = wbp + 4 * nW;                     // [4][H][H]

    auto cvt = [&](const float* s, __bf16* d, size_t n) {
        const int n4 = (int)(n / 4);
        cvt_f32_to_bf16<<<(n4 + 255) / 256, 256, 0, stream>>>(s, d, n4);
    };
    cvt(x, xb, nx);
    cvt(h_prev, hb, nh);
    cvt(W_i, wbp + 0 * nW, nW);
    cvt(W_f, wbp + 1 * nW, nW);
    cvt(W_o, wbp + 2 * nW, nW);
    cvt(W_g, wbp + 3 * nW, nW);
    cvt(U_i, ubp + 0 * nU, nU);
    cvt(U_f, ubp + 1 * nU, nU);
    cvt(U_o, ubp + 2 * nU, nU);
    cvt(U_g, ubp + 3 * nU, nU);

    dim3 grid(Bsz / BM, Hsz / BN);
    lstm_wmma_kernel<<<grid, THREADS, LDS_TOTAL, stream>>>(
        xb, hb, wbp, ubp,
        c_prev, boundary,
        b_Wi, b_Ui, b_Wf, b_Uf, b_Wo, b_Uo, b_Wg, b_Ug,
        W_b, b_Wb,
        out_h, out_c);
}
